// LocalAttention_12120397709565
// MI455X (gfx1250) — compile-verified, matrix operand order checked by reading, not run on hardware
//
#include <hip/hip_runtime.h>
#include <hip/hip_bf16.h>

typedef __attribute__((ext_vector_type(16))) _Float16 v16h;
typedef __attribute__((ext_vector_type(8)))  float    v8f;

#define S_LEN   2048
#define DHEAD   64
#define NHEAD   8
#define WIN     64          // half window
#define TQ      16          // queries per tile (WMMA M)
#define NKEYS   144         // TQ + 2*WIN window span of a tile
#define KPAD    160         // padded to multiple of 32 for PV K-dim
#define LSTRIDE 164         // LDS row stride in floats (164 dwords -> conflict-free)
#define WPB     4           // waves per block

__device__ __forceinline__ int clampi(int x, int lo, int hi) {
    return x < lo ? lo : (x > hi ? hi : x);
}

__global__ __launch_bounds__(WPB * 32)
void local_attn_wmma(const float* __restrict__ q,
                     const float* __restrict__ k,
                     const float* __restrict__ v,
                     float* __restrict__ out) {
    __shared__ float sP[WPB][TQ * LSTRIDE];   // scores -> probabilities
    __shared__ float sSum[WPB][TQ];           // softmax row sums

    const int lane   = threadIdx.x & 31;
    const int wave   = threadIdx.x >> 5;
    const int tile   = blockIdx.x * WPB + wave;        // 0 .. H*S/TQ-1
    const int h      = tile >> 7;                      // 128 tiles per head
    const int q0     = (tile & 127) * TQ;
    const int key0   = q0 - WIN;

    const float* Q = q   + (size_t)h * S_LEN * DHEAD;
    const float* K = k   + (size_t)h * S_LEN * DHEAD;
    const float* V = v   + (size_t)h * S_LEN * DHEAD;
    float*       O = out + (size_t)h * S_LEN * DHEAD;

    const int sub    = lane & 15;     // column / M%16 index
    const int halfid = lane >> 4;     // which half of the wave
    const int kb     = 8 * halfid;    // per-lane K-run base (A & B share mapping)
    float* P = sP[wave];

    // ---------------- Q fragments: A-layout, 2 K-chunks of 32 ----------------
    v16h aq[2];
    {
        const float* qrow = Q + (q0 + sub) * DHEAD;
#pragma unroll
        for (int c = 0; c < 2; ++c)
#pragma unroll
            for (int e = 0; e < 16; ++e) {
                int kk = (e < 8) ? (kb + e) : (16 + kb + (e - 8));
                aq[c][e] = (_Float16)qrow[c * 32 + kk];
            }
    }

    // ---------------- scores: S = (Q K^T) * scale, raw into LDS --------------
    const float scale = 0.125f;  // 1/sqrt(64)
    for (int kc = 0; kc < 9; ++kc) {
        int key  = key0 + kc * 16 + sub;
        const float* krow = K + clampi(key, 0, S_LEN - 1) * DHEAD;
        v8f acc = {};
#pragma unroll
        for (int c = 0; c < 2; ++c) {
            v16h bk;
#pragma unroll
            for (int e = 0; e < 16; ++e) {
                int kk = (e < 8) ? (kb + e) : (16 + kb + (e - 8));
                bk[e] = (_Float16)krow[c * 32 + kk];
            }
            acc = __builtin_amdgcn_wmma_f32_16x16x32_f16(
                    false, aq[c], false, bk, (short)0, acc, false, false);
        }
#pragma unroll
        for (int r = 0; r < 8; ++r) {
            int M = r + 8 * halfid;
            P[M * LSTRIDE + kc * 16 + sub] = acc[r] * scale;
        }
    }

    // zero-pad columns 144..159 so the PV K-loop can run to KPAD
#pragma unroll
    for (int t = 0; t < 8; ++t)
        P[sub * LSTRIDE + NKEYS + 8 * halfid + t] = 0.0f;

    // ---------------- masked softmax (fp32, in LDS) --------------------------
    {
        const int row   = sub;
        const int jlo   = max(row, WIN - q0);
        const int jhi   = min(min(row + 2 * WIN, NKEYS - 1), S_LEN - 1 - q0 + WIN);
        const int jbase = 72 * halfid;

        float m = -INFINITY;
        for (int t = 0; t < 72; ++t) {
            int j = jbase + t;
            float val = (j >= jlo && j <= jhi) ? P[row * LSTRIDE + j] : -INFINITY;
            m = fmaxf(m, val);
        }
        m = fmaxf(m, __shfl_xor(m, 16, 32));

        float ssum = 0.0f;
        for (int t = 0; t < 72; ++t) {
            int j = jbase + t;
            float val = (j >= jlo && j <= jhi) ? P[row * LSTRIDE + j] : -INFINITY;
            float e = __expf(val - m);          // masked -> exp(-inf)=0
            P[row * LSTRIDE + j] = e;
            ssum += e;
        }
        ssum += __shfl_xor(ssum, 16, 32);
        if (halfid == 0) sSum[wave][row] = ssum;
    }

    // per-lane inverse row sums for the output rows this lane owns
    float isum[8];
#pragma unroll
    for (int r = 0; r < 8; ++r)
        isum[r] = 1.0f / sSum[wave][r + 8 * halfid];

    // ---------------- O = P * V (K-dim padded to 160) ------------------------
    for (int nb = 0; nb < 4; ++nb) {
        v8f acc = {};
        for (int st = 0; st < 5; ++st) {
            v16h ap, bv;
#pragma unroll
            for (int e = 0; e < 16; ++e) {
                int kk = (e < 8) ? (kb + e) : (16 + kb + (e - 8));
                int j  = st * 32 + kk;
                ap[e]  = (_Float16)P[sub * LSTRIDE + j];
                int key = clampi(key0 + j, 0, S_LEN - 1);   // prob is 0 where clamped
                bv[e]  = (_Float16)V[key * DHEAD + nb * 16 + sub];
            }
            acc = __builtin_amdgcn_wmma_f32_16x16x32_f16(
                    false, ap, false, bv, (short)0, acc, false, false);
        }
#pragma unroll
        for (int r = 0; r < 8; ++r) {
            int M = r + 8 * halfid;
            O[(q0 + M) * DHEAD + nb * 16 + sub] = acc[r] * isum[r];
        }
    }
}

extern "C" void kernel_launch(void* const* d_in, const int* in_sizes, int n_in,
                              void* d_out, int out_size, void* d_ws, size_t ws_size,
                              hipStream_t stream) {
    (void)in_sizes; (void)n_in; (void)d_ws; (void)ws_size; (void)out_size;
    const float* q = (const float*)d_in[0];
    const float* k = (const float*)d_in[1];
    const float* v = (const float*)d_in[2];
    float* out = (float*)d_out;

    const int tiles  = NHEAD * (S_LEN / TQ);   // 1024
    const int blocks = tiles / WPB;            // 256
    local_attn_wmma<<<blocks, WPB * 32, 0, stream>>>(q, k, v, out);
}
